// InteractionBlock_old_32796370272381
// MI455X (gfx1250) — matrix operand order repaired and checked by reading, not compiled, not run on hardware
//
#include <hip/hip_runtime.h>
#include <hip/hip_bf16.h>
#include <math.h>

// ---------------------------------------------------------------------------
// SchNet InteractionBlock for MI455X (gfx1250, wave32, WMMA bf16/f32-acc)
// Weights pre-swizzled to WMMA-fragment order so B operands are single
// contiguous 32B LDS reads (2x ds_load_b128); A operands are 2x 16B reads.
// ---------------------------------------------------------------------------

typedef __attribute__((ext_vector_type(16))) __bf16 v16bf;
typedef __attribute__((ext_vector_type(8)))  __bf16 v8bf;
typedef __attribute__((ext_vector_type(4)))  __bf16 v4bf;
typedef __attribute__((ext_vector_type(8)))  float  v8f;

#define N_NODES 100000
#define N_EDGES 1600000
#define HDIM    128
#define DDIM    50
#define PI_OVER_CUTOFF 0.39269908169872414f   // pi/8
#define LN2F    0.6931471805599453f

// ---- workspace layout (bytes) ----
static constexpr size_t OFF_W1  = 0;                       // 2 kchunks x 8 nt x 32 lanes x 16 (swizzled, padded K->64)
static constexpr size_t OFF_W2  = OFF_W1 + 64 * 128 * 2;   // 4 kchunks x 8 nt x 32 x 16
static constexpr size_t OFF_L1  = OFF_W2 + 128 * 128 * 2;
static constexpr size_t OFF_L2  = OFF_L1 + 128 * 128 * 2;
static constexpr size_t OFF_LW  = OFF_L2 + 128 * 128 * 2;
static constexpr size_t OFF_CS1 = OFF_LW + 128 * 128 * 2;  // col sums / sumsq (contiguous 4x128 f32)
static constexpr size_t OFF_CQ1 = OFF_CS1 + 128 * 4;
static constexpr size_t OFF_CS2 = OFF_CQ1 + 128 * 4;
static constexpr size_t OFF_CQ2 = OFF_CS2 + 128 * 4;
static constexpr size_t OFF_A1A = OFF_CQ2 + 128 * 4;       // folded bn1 affine (x2 doubling)
static constexpr size_t OFF_A1B = OFF_A1A + 128 * 4;
static constexpr size_t OFF_A2A = OFF_A1B + 128 * 4;
static constexpr size_t OFF_A2B = OFF_A2A + 128 * 4;
static constexpr size_t OFF_H1  = 151552;                  // N x H f32
static constexpr size_t OFF_AGG = OFF_H1  + (size_t)N_NODES * HDIM * 4;
static constexpr size_t OFF_Y1  = OFF_AGG + (size_t)N_NODES * HDIM * 4;
static constexpr size_t OFF_Y2  = OFF_Y1  + (size_t)N_NODES * HDIM * 4;

// ---- helpers ----
__device__ __forceinline__ float sspf(float x) {
  // softplus(x) - ln(2), numerically stable
  float sp = fmaxf(x, 0.0f) + log1pf(__expf(-fabsf(x)));
  return sp - LN2F;
}

// A fragment (16x32 bf16, ISA 7.12.2) from a row-major LDS tile:
// elems 0..7  = K(kb + half*8 + 0..7)      -> one contiguous 16B run
// elems 8..15 = K(kb + 16 + half*8 + 0..7) -> one contiguous 16B run
__device__ __forceinline__ v16bf load_a(const __bf16* S, int srow, int kb,
                                        int half, int lane) {
  const v8bf lo = *(const v8bf*)(S + lane * srow + kb + half * 8);
  const v8bf hi = *(const v8bf*)(S + lane * srow + kb + 16 + half * 8);
  return __builtin_shufflevector(lo, hi, 0, 1, 2, 3, 4, 5, 6, 7, 8, 9, 10, 11,
                                 12, 13, 14, 15);
}

// B fragment from fragment-order swizzled weights: one contiguous 32B read.
__device__ __forceinline__ v16bf load_b_swz(const __bf16* Wswz, int kc, int nt,
                                            int lid) {
  return *(const v16bf*)(Wswz + (((kc * 8 + nt) * 32) + lid) * 16);
}

__device__ __forceinline__ v8f wmma_bf16(v16bf a, v16bf b, v8f c) {
  return __builtin_amdgcn_wmma_f32_16x16x32_bf16(false, a, false, b,
                                                 (short)0, c, false, false);
}

// ---------------------------------------------------------------------------
// kernel: zero agg + bn statistics
// ---------------------------------------------------------------------------
__global__ __launch_bounds__(256) void zero_kernel(float* __restrict__ agg,
                                                   float* __restrict__ stats) {
  size_t i = (size_t)blockIdx.x * blockDim.x + threadIdx.x;
  size_t stride = (size_t)gridDim.x * blockDim.x;
  for (size_t j = i; j < (size_t)N_NODES * HDIM; j += stride) agg[j] = 0.0f;
  if (i < 512) stats[i] = 0.0f;
}

// ---------------------------------------------------------------------------
// kernel: convert fp32 weights -> bf16 in WMMA B-fragment order.
// dst[((kc*8 + nt)*32 + l)*16 + i] = W[(kc*32 + (i>>3)*16 + (l>>4)*8 + (i&7))*128 + nt*16 + (l&15)]
// (mlp_w1 zero-padded K 50 -> 64, i.e. kc in {0,1})
// ---------------------------------------------------------------------------
__global__ __launch_bounds__(256) void cvt_weights_kernel(
    const float* __restrict__ w1, const float* __restrict__ w2,
    const float* __restrict__ l1, const float* __restrict__ l2,
    const float* __restrict__ lw, __bf16* __restrict__ W1bf,
    __bf16* __restrict__ W2bf, __bf16* __restrict__ L1bf,
    __bf16* __restrict__ L2bf, __bf16* __restrict__ LWbf) {
  int idx = blockIdx.x * blockDim.x + threadIdx.x;
  int i = idx & 15;
  int l = (idx >> 4) & 31;
  int nt = (idx >> 9) & 7;
  int kc = idx >> 12;
  int k = kc * 32 + ((i >> 3) << 4) + ((l >> 4) << 3) + (i & 7);
  int n = nt * 16 + (l & 15);
  if (idx < 64 * 128) {  // kc in {0,1}
    W1bf[idx] = (__bf16)((k < DDIM) ? w1[k * 128 + n] : 0.0f);
  }
  if (idx < 128 * 128) {  // kc in {0..3}
    W2bf[idx] = (__bf16)w2[k * 128 + n];
    L1bf[idx] = (__bf16)l1[k * 128 + n];
    L2bf[idx] = (__bf16)l2[k * 128 + n];
    LWbf[idx] = (__bf16)lw[k * 128 + n];
  }
}

// ---------------------------------------------------------------------------
// kernel: row-tile GEMM  Y[t] = f(X[t]) @ W + bias   (16 rows/wave, 2 waves)
//   mode 0: f = identity, no stats
//   mode 1: f = identity, accumulate per-column sum/sumsq of Y
//   mode 2: f(x) = ssp(affA[c]*x + affB[c]), accumulate stats
// ---------------------------------------------------------------------------
__global__ __launch_bounds__(64) void gemm_rowtile_kernel(
    const float* __restrict__ X, const __bf16* __restrict__ Wbf,
    const float* __restrict__ bias, float* __restrict__ Y,
    const float* __restrict__ affA, const float* __restrict__ affB,
    float* __restrict__ colsum, float* __restrict__ colsumsq,
    int ntiles, int mode) {
  __shared__ __attribute__((aligned(16))) __bf16 Ws[128 * 128];  // swizzled
  __shared__ float bs[128];
  __shared__ float aA[128];
  __shared__ float aB[128];
  __shared__ __attribute__((aligned(16))) __bf16 Xs[2][16 * 128];

  const int tid = threadIdx.x;
  {  // weights are already swizzled -> flat 16B copies
    const uint4* src = (const uint4*)Wbf;
    uint4* dst = (uint4*)Ws;
    for (int i = tid; i < (128 * 128) / 8; i += 64) dst[i] = src[i];
  }
  for (int i = tid; i < 128; i += 64) {
    bs[i] = bias[i];
    if (mode == 2) { aA[i] = affA[i]; aB[i] = affB[i]; }
  }
  __syncthreads();

  const int wave = tid >> 5, lid = tid & 31;
  const int half = lid >> 4, lane = lid & 15;

  for (int t = blockIdx.x * 2 + wave; t < ntiles; t += gridDim.x * 2) {
    const float* Xrow = X + (size_t)t * 16 * 128;
    __bf16* xs = Xs[wave];
    // stage 16x128 tile as bf16: float4 load -> 4xbf16 pack -> 8B LDS store
    for (int j = lid; j < (16 * 128) / 4; j += 32) {
      float4 x4 = *(const float4*)(Xrow + j * 4);
      if (mode == 2) {
        int c = (j * 4) & 127;
        x4.x = sspf(aA[c + 0] * x4.x + aB[c + 0]);
        x4.y = sspf(aA[c + 1] * x4.y + aB[c + 1]);
        x4.z = sspf(aA[c + 2] * x4.z + aB[c + 2]);
        x4.w = sspf(aA[c + 3] * x4.w + aB[c + 3]);
      }
      v4bf p;
      p[0] = (__bf16)x4.x; p[1] = (__bf16)x4.y;
      p[2] = (__bf16)x4.z; p[3] = (__bf16)x4.w;
      *(v4bf*)(xs + j * 4) = p;
    }
    // A fragments held in registers across all 8 n-tiles
    v16bf afrag[4];
#pragma unroll
    for (int kc = 0; kc < 4; ++kc)
      afrag[kc] = load_a(xs, 128, kc * 32, half, lane);
#pragma unroll 1
    for (int nt = 0; nt < 8; ++nt) {
      v8f acc = {0, 0, 0, 0, 0, 0, 0, 0};
#pragma unroll
      for (int kc = 0; kc < 4; ++kc)
        acc = wmma_bf16(afrag[kc], load_b_swz(Ws, kc, nt, lid), acc);
      int col = nt * 16 + lane;
      float bb = bs[col];
      float s = 0.0f, sq = 0.0f;
#pragma unroll
      for (int r = 0; r < 8; ++r) {
        float y = acc[r] + bb;
        Y[((size_t)t * 16 + half * 8 + r) * 128 + col] = y;
        s += y;
        sq += y * y;
      }
      if (mode) {
        s  += __shfl_xor(s, 16, 32);   // lanes l and l+16 share a column
        sq += __shfl_xor(sq, 16, 32);
        if (lid < 16) {
          atomicAdd(&colsum[col], s);
          atomicAdd(&colsumsq[col], sq);
        }
      }
    }
  }
}

// ---------------------------------------------------------------------------
// kernel: fused edge path — filter MLP (two WMMA GEMMs) + cutoff +
//         gather h1[src] * w + atomic scatter-add into agg[dst]
// ---------------------------------------------------------------------------
__global__ __launch_bounds__(64) void edge_kernel(
    const float* __restrict__ ef, const float* __restrict__ dist,
    const int* __restrict__ eidx, const __bf16* __restrict__ W1g,
    const __bf16* __restrict__ W2g, const float* __restrict__ b1,
    const float* __restrict__ b2, const float* __restrict__ h1,
    float* __restrict__ agg) {
  __shared__ __attribute__((aligned(16))) __bf16 W1s[64 * 128];    // swizzled
  __shared__ __attribute__((aligned(16))) __bf16 W2s[128 * 128];   // swizzled
  __shared__ float b1s[128];
  __shared__ float b2s[128];
  __shared__ __attribute__((aligned(16))) __bf16 Ab[2][16 * 64];
  __shared__ __attribute__((aligned(16))) __bf16 Tb[2][16 * 128];

  const int tid = threadIdx.x;
  {
    const uint4* s1 = (const uint4*)W1g;
    uint4* d1 = (uint4*)W1s;
    for (int i = tid; i < (64 * 128) / 8; i += 64) d1[i] = s1[i];
    const uint4* s2 = (const uint4*)W2g;
    uint4* d2 = (uint4*)W2s;
    for (int i = tid; i < (128 * 128) / 8; i += 64) d2[i] = s2[i];
  }
  for (int i = tid; i < 128; i += 64) { b1s[i] = b1[i]; b2s[i] = b2[i]; }
  __syncthreads();

  const int wave = tid >> 5, lid = tid & 31;
  const int half = lid >> 4, lane = lid & 15;
  const int NT = N_EDGES / 16;
  const long tile_step = (long)gridDim.x * 2;

  for (long t = (long)blockIdx.x * 2 + wave; t < NT; t += tile_step) {
    const long e0 = t * 16;
    // prefetch next tile's edge features into the cache hierarchy
    long enext = e0 + tile_step * 16;
    if (enext < N_EDGES && lid == 0)
      __builtin_prefetch(&ef[enext * DDIM], 0, 1);

    // ---- stage 16x64 bf16 A-tile (K padded 50 -> 64) ----
    __bf16* ab = Ab[wave];
    for (int i = lid; i < 16 * 64; i += 32) {
      int r = i >> 6, k = i & 63;
      float v = (k < DDIM) ? ef[(e0 + r) * DDIM + k] : 0.0f;
      ab[i] = (__bf16)v;
    }

    // ---- GEMM1: t = ssp(A @ W1 + b1)  -> Tb (bf16) ----
    __bf16* tb = Tb[wave];
    {
      v16bf a0 = load_a(ab, 64, 0, half, lane);
      v16bf a1 = load_a(ab, 64, 32, half, lane);
#pragma unroll 1
      for (int nt = 0; nt < 8; ++nt) {
        v8f acc = {0, 0, 0, 0, 0, 0, 0, 0};
        acc = wmma_bf16(a0, load_b_swz(W1s, 0, nt, lid), acc);
        acc = wmma_bf16(a1, load_b_swz(W1s, 1, nt, lid), acc);
        int col = nt * 16 + lane;
        float bb = b1s[col];
#pragma unroll
        for (int r = 0; r < 8; ++r)
          tb[(half * 8 + r) * 128 + col] = (__bf16)sspf(acc[r] + bb);
      }
    }

    // ---- per-row cutoff + edge endpoints ----
    float cut[8];
    int si[8], di[8];
#pragma unroll
    for (int r = 0; r < 8; ++r) {
      long e = e0 + half * 8 + r;
      cut[r] = 0.5f * (__cosf(dist[e] * PI_OVER_CUTOFF) + 1.0f);
      si[r] = eidx[e];            // src
      di[r] = eidx[N_EDGES + e];  // dst
    }

    // ---- GEMM2: w = (t @ W2 + b2) * cut ; gather/scatter ----
    v16bf afrag[4];
#pragma unroll
    for (int kc = 0; kc < 4; ++kc)
      afrag[kc] = load_a(tb, 128, kc * 32, half, lane);
#pragma unroll 1
    for (int nt = 0; nt < 8; ++nt) {
      v8f acc = {0, 0, 0, 0, 0, 0, 0, 0};
#pragma unroll
      for (int kc = 0; kc < 4; ++kc)
        acc = wmma_bf16(afrag[kc], load_b_swz(W2s, kc, nt, lid), acc);
      int col = nt * 16 + lane;
      float bb = b2s[col];
#pragma unroll
      for (int r = 0; r < 8; ++r) {
        float w = (acc[r] + bb) * cut[r];
        float msg = h1[(size_t)si[r] * HDIM + col] * w;
        atomicAdd(&agg[(size_t)di[r] * HDIM + col], msg);
      }
    }
  }
}

// ---------------------------------------------------------------------------
// kernel: finalize batchnorm affine  (scale folds the `h+h` doubling for bn1)
// ---------------------------------------------------------------------------
__global__ __launch_bounds__(128) void bn_finalize_kernel(
    const float* __restrict__ cs, const float* __restrict__ cq,
    const float* __restrict__ g, const float* __restrict__ b,
    float* __restrict__ aA, float* __restrict__ aB, float scale) {
  int c = threadIdx.x;
  float mu  = cs[c] * (1.0f / N_NODES);
  float var = cq[c] * (1.0f / N_NODES) - mu * mu;
  float a = g[c] * rsqrtf(var + 1e-5f);
  aA[c] = scale * a;
  aB[c] = scale * (b[c] - mu * a);
}

// ---------------------------------------------------------------------------
// kernel: out = bn2(y2) + node_feats
// ---------------------------------------------------------------------------
__global__ __launch_bounds__(256) void final_kernel(
    const float* __restrict__ y2, const float* __restrict__ aA,
    const float* __restrict__ aB, const float* __restrict__ orig,
    float* __restrict__ out) {
  size_t i = (size_t)blockIdx.x * blockDim.x + threadIdx.x;
  size_t stride = (size_t)gridDim.x * blockDim.x;
  for (size_t j = i; j < (size_t)N_NODES * HDIM; j += stride) {
    int c = (int)(j & 127);
    out[j] = aA[c] * y2[j] + aB[c] + orig[j];
  }
}

// ---------------------------------------------------------------------------
extern "C" void kernel_launch(void* const* d_in, const int* in_sizes, int n_in,
                              void* d_out, int out_size, void* d_ws,
                              size_t ws_size, hipStream_t stream) {
  (void)in_sizes; (void)n_in; (void)out_size; (void)ws_size;
  const float* node_feats = (const float*)d_in[0];
  const float* edge_feats = (const float*)d_in[1];
  const float* distances  = (const float*)d_in[2];
  const int*   edge_index = (const int*)d_in[3];
  const float* mlp_w1 = (const float*)d_in[4];
  const float* mlp_b1 = (const float*)d_in[5];
  const float* mlp_w2 = (const float*)d_in[6];
  const float* mlp_b2 = (const float*)d_in[7];
  const float* lin1_w = (const float*)d_in[8];
  const float* lin1_b = (const float*)d_in[9];
  const float* lin2_w = (const float*)d_in[10];
  const float* lin2_b = (const float*)d_in[11];
  const float* bn1_g  = (const float*)d_in[12];
  const float* bn1_b  = (const float*)d_in[13];
  const float* lin_w  = (const float*)d_in[14];
  const float* lin_b  = (const float*)d_in[15];
  const float* bn2_g  = (const float*)d_in[16];
  const float* bn2_b  = (const float*)d_in[17];

  char* ws = (char*)d_ws;
  __bf16* W1bf = (__bf16*)(ws + OFF_W1);
  __bf16* W2bf = (__bf16*)(ws + OFF_W2);
  __bf16* L1bf = (__bf16*)(ws + OFF_L1);
  __bf16* L2bf = (__bf16*)(ws + OFF_L2);
  __bf16* LWbf = (__bf16*)(ws + OFF_LW);
  float* cs1 = (float*)(ws + OFF_CS1);
  float* cq1 = (float*)(ws + OFF_CQ1);
  float* cs2 = (float*)(ws + OFF_CS2);
  float* cq2 = (float*)(ws + OFF_CQ2);
  float* a1A = (float*)(ws + OFF_A1A);
  float* a1B = (float*)(ws + OFF_A1B);
  float* a2A = (float*)(ws + OFF_A2A);
  float* a2B = (float*)(ws + OFF_A2B);
  float* h1  = (float*)(ws + OFF_H1);
  float* agg = (float*)(ws + OFF_AGG);
  float* y1  = (float*)(ws + OFF_Y1);
  float* y2  = (float*)(ws + OFF_Y2);

  const int ntiles_n = N_NODES / 16;  // 6250

  zero_kernel<<<2048, 256, 0, stream>>>(agg, cs1);
  cvt_weights_kernel<<<64, 256, 0, stream>>>(mlp_w1, mlp_w2, lin1_w, lin2_w,
                                             lin_w, W1bf, W2bf, L1bf, L2bf,
                                             LWbf);
  // h1 = node_feats @ lin1_w + lin1_b
  gemm_rowtile_kernel<<<1024, 64, 0, stream>>>(node_feats, L1bf, lin1_b, h1,
                                               nullptr, nullptr, nullptr,
                                               nullptr, ntiles_n, 0);
  // fused edge filter + message passing
  edge_kernel<<<4096, 64, 0, stream>>>(edge_feats, distances, edge_index, W1bf,
                                       W2bf, mlp_b1, mlp_b2, h1, agg);
  // y1 = agg @ lin2_w + lin2_b  (+ bn1 stats)
  gemm_rowtile_kernel<<<1024, 64, 0, stream>>>(agg, L2bf, lin2_b, y1, nullptr,
                                               nullptr, cs1, cq1, ntiles_n, 1);
  bn_finalize_kernel<<<1, 128, 0, stream>>>(cs1, cq1, bn1_g, bn1_b, a1A, a1B,
                                            2.0f);  // folds h = h + h
  // y2 = ssp(bn1'(y1)) @ lin_w + lin_b  (+ bn2 stats)
  gemm_rowtile_kernel<<<1024, 64, 0, stream>>>(y1, LWbf, lin_b, y2, a1A, a1B,
                                               cs2, cq2, ntiles_n, 2);
  bn_finalize_kernel<<<1, 128, 0, stream>>>(cs2, cq2, bn2_g, bn2_b, a2A, a2B,
                                            1.0f);
  final_kernel<<<2048, 256, 0, stream>>>(y2, a2A, a2B, node_feats,
                                         (float*)d_out);
}
